// Attention_30133490549270
// MI455X (gfx1250) — matrix operand order repaired and checked by reading, not compiled, not run on hardware
//
#include <hip/hip_runtime.h>
#include <hip/hip_bf16.h>
#include <math.h>

// ---------------- problem constants ----------------
#define B_  2
#define L_  2048
#define P_  512
#define D_  4096
#define H_  32
#define KH_ 8
#define HD_ 128
#define S_  (P_ + L_)      // 2560
#define G_  (H_ / KH_)     // 4

typedef __attribute__((ext_vector_type(16))) _Float16 v16h;
typedef __attribute__((ext_vector_type(8)))  float    v8f;

typedef unsigned int       u32;
typedef unsigned long long u64;
typedef unsigned int u32x4 __attribute__((ext_vector_type(4)));
typedef int          i32x8 __attribute__((ext_vector_type(8)));
typedef int          i32x4 __attribute__((ext_vector_type(4)));

static __device__ inline v8f zero8() {
    v8f z;
#pragma unroll
    for (int i = 0; i < 8; ++i) z[i] = 0.0f;
    return z;
}

// Low 32 bits of a generic (flat) shared-memory address = LDS byte offset.
static __device__ inline u32 lds_off_of(const void* p) {
    return (u32)(u64)p;
}

// ---------------- TDM 2D tile load: global -> LDS ----------------
// D# per CDNA5 ISA 8.3/8.4: group0 = {count/type/addr}, group1 = dims/strides.
// data_size = 2 bytes (f16). tile rows are packed contiguously in LDS.
static __device__ inline void tdm_load_2d_f16(const void* gptr, u32 lds_byte_off,
                                              u32 tile_d0, u32 tile_d1,
                                              u32 tensor_d0, u32 tensor_d1,
                                              u32 stride_elems) {
    u64 ga = (u64)gptr;
    u32x4 g0;
    g0.x = 1u;                                   // count=1 (valid), user mode
    g0.y = lds_byte_off;                         // lds_addr
    g0.z = (u32)ga;                              // global_addr[31:0]
    g0.w = (u32)((ga >> 32) & 0x01FFFFFFu) | 0x80000000u;   // addr[56:32] | type=2
    i32x8 g1;
    g1[0] = (int)(1u << 16);                     // wg_mask=0, data_size=1 (2B)
    g1[1] = (int)((tensor_d0 & 0xFFFFu) << 16);  // atomic_barrier=0 | dim0.lo16
    g1[2] = (int)((tensor_d0 >> 16) | ((tensor_d1 & 0xFFFFu) << 16));
    g1[3] = (int)((tensor_d1 >> 16) | (tile_d0 << 16));     // dim1.hi16 | tile_dim0
    g1[4] = (int)(tile_d1 & 0xFFFFu);            // tile_dim1 | tile_dim2=0
    g1[5] = (int)stride_elems;                   // tensor_dim0_stride[31:0]
    g1[6] = 0;                                   // stride hi16 | dim1_stride lo16
    g1[7] = 0;
    i32x4 z4 = {0, 0, 0, 0};
#if __has_include(<hip/amd_detail/amd_gfx1250_TDM.h>)
    i32x8 z8 = {0, 0, 0, 0, 0, 0, 0, 0};
    __builtin_amdgcn_tensor_load_to_lds(g0, g1, z4, z4, z8, 0);
#else
    __builtin_amdgcn_tensor_load_to_lds(g0, g1, z4, z4, 0);
#endif
}

// ---------------- fp32 -> fp16 conversion (4 elems/thread) ----------------
__global__ void cvt_f32_to_f16(const float* __restrict__ src,
                               _Float16* __restrict__ dst, int n4) {
    int i = blockIdx.x * blockDim.x + threadIdx.x;
    if (i < n4) {
        float4 v = reinterpret_cast<const float4*>(src)[i];
        dst[4 * i + 0] = (_Float16)v.x;
        dst[4 * i + 1] = (_Float16)v.y;
        dst[4 * i + 2] = (_Float16)v.z;
        dst[4 * i + 3] = (_Float16)v.w;
    }
}

// past_k/past_v  [B,KH,P,HD] fp32  ->  k_all/v_all [B,KH,S,HD] f16 at s in [0,P)
__global__ void pack_past(const float* __restrict__ past,
                          _Float16* __restrict__ dst, int n) {
    int i = blockIdx.x * blockDim.x + threadIdx.x;
    if (i >= n) return;
    int hd = i & (HD_ - 1);
    int t  = i >> 7;
    int p  = t % P_;
    int bk = t / P_;
    dst[((size_t)bk * S_ + p) * HD_ + hd] = (_Float16)past[i];
}

// proj [B,L,KH,HD] f16 -> k_all/v_all [B,KH,S,HD] f16 at s = P + l
__global__ void pack_new(const _Float16* __restrict__ proj,
                         _Float16* __restrict__ dst, int n) {
    int i = blockIdx.x * blockDim.x + threadIdx.x;
    if (i >= n) return;
    int hd = i & (HD_ - 1);
    int t  = i >> 7;
    int kh = t % KH_;
    int l  = (t / KH_) % L_;
    int b  = t / (KH_ * L_);
    dst[(((size_t)b * KH_ + kh) * S_ + (P_ + l)) * HD_ + hd] = proj[i];
}

// ---------------- tiled WMMA GEMM with TDM double-buffered staging ----------
// C[M,N] = A[M,K] * B[K,N], A/B f16 row-major. BM=BN=128, BK=64.
// 256 threads = 8 waves; wave (wy,wx): 64 rows x 32 cols; 4x2 WMMA tiles each.
// Wave 0 drives the Tensor Data Mover; TENSORcnt-based software pipeline.
template <typename OutT>
__global__ __launch_bounds__(256)
void gemm_wmma_f16(const _Float16* __restrict__ A,
                   const _Float16* __restrict__ Bm,
                   OutT* __restrict__ C, int M, int N, int K) {
    constexpr int BM = 128, BN = 128, BK = 64;
    __shared__ __align__(16) _Float16 sA[2][BM * BK];   // 2 x 16 KB
    __shared__ __align__(16) _Float16 sB[2][BK * BN];   // 2 x 16 KB

    const int tid   = threadIdx.x;
    const int lane  = tid & 31;
    const int w     = tid >> 5;
    const int wy    = w & 1;
    const int wx    = w >> 1;
    const int row16 = lane & 15;
    const int hi    = lane >> 4;

    const int m_blk = blockIdx.y * BM;
    const int n_blk = blockIdx.x * BN;

    v8f acc[4][2];
#pragma unroll
    for (int i = 0; i < 4; ++i)
#pragma unroll
        for (int j = 0; j < 2; ++j) acc[i][j] = zero8();

    // TDM issue: A tile (BM rows x BK) and B tile (BK rows x BN)
    auto issue_tiles = [&](int buf, int k0) {
        tdm_load_2d_f16(A + (size_t)m_blk * K + k0, lds_off_of(&sA[buf][0]),
                        BK, BM, (u32)(K - k0), (u32)(M - m_blk), (u32)K);
        tdm_load_2d_f16(Bm + (size_t)k0 * N + n_blk, lds_off_of(&sB[buf][0]),
                        BN, BK, (u32)(N - n_blk), (u32)(K - k0), (u32)N);
    };

    if (w == 0) issue_tiles(0, 0);

    for (int k0 = 0; k0 < K; k0 += BK) {
        const int cur = (k0 / BK) & 1;
        if (w == 0) {
            if (k0 + BK < K) {
                issue_tiles(cur ^ 1, k0 + BK);
                __builtin_amdgcn_s_wait_tensorcnt(2);   // current tiles done; next 2 in flight
            } else {
                __builtin_amdgcn_s_wait_tensorcnt(0);   // drain
            }
        }
        __syncthreads();   // current tiles resident in LDS for all waves

#pragma unroll
        for (int ks = 0; ks < BK / 32; ++ks) {
            v16h af[4];
#pragma unroll
            for (int i = 0; i < 4; ++i) {
                const _Float16* ap = &sA[cur][(wy * 64 + i * 16 + row16) * BK + ks * 32 + hi * 16];
#pragma unroll
                for (int j = 0; j < 8; ++j) af[i][j] = ap[j];
#pragma unroll
                for (int j = 0; j < 8; ++j) af[i][8 + j] = ap[8 + j];
            }
            v16h bf[2];
#pragma unroll
            for (int j2 = 0; j2 < 2; ++j2) {
                const _Float16* bp = &sB[cur][(ks * 32 + hi * 16) * BN + wx * 32 + j2 * 16 + row16];
#pragma unroll
                for (int j = 0; j < 16; ++j) bf[j2][j] = bp[j * BN];
            }
#pragma unroll
            for (int i = 0; i < 4; ++i)
#pragma unroll
                for (int j2 = 0; j2 < 2; ++j2)
                    acc[i][j2] = __builtin_amdgcn_wmma_f32_16x16x32_f16(
                        false, af[i], false, bf[j2], (short)0, acc[i][j2], false, false);
        }
        __syncthreads();   // all waves done reading before TDM overwrites this buffer
    }

#pragma unroll
    for (int i = 0; i < 4; ++i)
#pragma unroll
        for (int j2 = 0; j2 < 2; ++j2)
#pragma unroll
            for (int e = 0; e < 8; ++e) {
                int r = m_blk + wy * 64 + i * 16 + hi * 8 + e;
                int c = n_blk + wx * 32 + j2 * 16 + row16;
                C[(size_t)r * N + c] = (OutT)acc[i][j2][e];
            }
}

// ---------------- flash attention core ----------------
// grid.x = B*H, grid.y = L/64.  Block = 256 threads (8 waves).
// K/V streamed in 32-key chunks via TDM (double-buffered); online softmax;
// WMMA for QK^T and P*V.
__global__ __launch_bounds__(256)
void attn_flash(const _Float16* __restrict__ q16,    // [B, L, H*HD]
                const _Float16* __restrict__ k_all,  // [B, KH, S, HD]
                const _Float16* __restrict__ v_all,  // [B, KH, S, HD]
                const float* __restrict__ mask,      // [L, S]
                _Float16* __restrict__ out) {        // [B, L, H*HD]
    constexpr int QT = 64;   // query tile
    constexpr int CK = 32;   // key chunk
    __shared__ __align__(16) _Float16 sQ[QT * HD_];        // 16 KB
    __shared__ __align__(16) _Float16 sK[2][CK * HD_];     // 16 KB
    __shared__ __align__(16) _Float16 sV[2][CK * HD_];     // 16 KB
    __shared__ float     sS[QT * CK];                      //  8 KB
    __shared__ _Float16  sP[QT * CK];                      //  4 KB
    __shared__ float sM[QT], sL[QT], sAl[QT];

    const int tid   = threadIdx.x;
    const int lane  = tid & 31;
    const int w     = tid >> 5;
    const int row16 = lane & 15;
    const int hi    = lane >> 4;

    const int bh = blockIdx.x;
    const int b  = bh / H_;
    const int h  = bh % H_;
    const int kh = h / G_;
    const int q0 = blockIdx.y * QT;

    const float scale = 0.08838834764831845f;  // 1/sqrt(128)

    const _Float16* kbase = k_all + ((size_t)(b * KH_ + kh)) * (size_t)S_ * HD_;
    const _Float16* vbase = v_all + ((size_t)(b * KH_ + kh)) * (size_t)S_ * HD_;

    auto issue_kv = [&](int buf, int s0) {
        tdm_load_2d_f16(kbase + (size_t)s0 * HD_, lds_off_of(&sK[buf][0]),
                        HD_, CK, HD_, (u32)(S_ - s0), HD_);
        tdm_load_2d_f16(vbase + (size_t)s0 * HD_, lds_off_of(&sV[buf][0]),
                        HD_, CK, HD_, (u32)(S_ - s0), HD_);
    };

    if (w == 0) issue_kv(0, 0);   // prime the pipeline

    // ---- load Q tile (64 x 128) through VGPRs (one-time) ----
    {
        uint4* lQ = reinterpret_cast<uint4*>(sQ);
#pragma unroll
        for (int it = 0; it < 4; ++it) {
            int idx = tid + it * 256;      // 0..1023, 16 uint4 per row
            int r   = idx >> 4;
            int cs  = idx & 15;
            const uint4* g = reinterpret_cast<const uint4*>(
                q16 + ((size_t)(b * L_ + q0 + r)) * (size_t)D_ + h * HD_);
            lQ[idx] = g[cs];
        }
    }
    if (tid < QT) { sM[tid] = -1e30f; sL[tid] = 0.0f; }

    v8f oacc[4];
#pragma unroll
    for (int jt = 0; jt < 4; ++jt) oacc[jt] = zero8();
    const int wr = w >> 1;       // 0..3 (row group)
    const int wc = w & 1;        // 0..1 (col half)

    const int kend    = P_ + q0 + QT;   // causal bound, multiple of CK
    const int nchunks = kend / CK;

    for (int c = 0; c < nchunks; ++c) {
        const int s0  = c * CK;
        const int cur = c & 1;
        if (w == 0) {
            if (c + 1 < nchunks) {
                issue_kv(cur ^ 1, s0 + CK);
                __builtin_amdgcn_s_wait_tensorcnt(2);   // current chunk done
            } else {
                __builtin_amdgcn_s_wait_tensorcnt(0);   // drain
            }
        }
        __syncthreads();   // chunk resident (also covers Q-tile/init on c==0)

        // ---- scores: 64x32 = 4x2 tiles of 16x16, one per wave ----
        {
            const int tr = w >> 1;
            const int tc = w & 1;
            v8f sacc = zero8();
#pragma unroll
            for (int ks = 0; ks < HD_ / 32; ++ks) {
                v16h af;
                const _Float16* ap = &sQ[(tr * 16 + row16) * HD_ + ks * 32 + hi * 16];
#pragma unroll
                for (int j = 0; j < 8; ++j) af[j] = ap[j];
#pragma unroll
                for (int j = 0; j < 8; ++j) af[8 + j] = ap[8 + j];
                v16h bf;   // B[kk][n] = K[key n][hd kk]
                const _Float16* bp = &sK[cur][(tc * 16 + row16) * HD_ + ks * 32 + hi * 16];
#pragma unroll
                for (int j = 0; j < 16; ++j) bf[j] = bp[j];
                sacc = __builtin_amdgcn_wmma_f32_16x16x32_f16(
                    false, af, false, bf, (short)0, sacc, false, false);
            }
#pragma unroll
            for (int e = 0; e < 8; ++e) {
                int qr = tr * 16 + hi * 8 + e;
                int kc = tc * 16 + row16;
                float mval = mask[(size_t)(q0 + qr) * S_ + (s0 + kc)];
                sS[qr * CK + kc] = sacc[e] * scale + mval;
            }
        }
        __syncthreads();

        // ---- online softmax update (one thread per query row) ----
        if (tid < QT) {
            int r = tid;
            float mrow = -1e30f;
#pragma unroll
            for (int j = 0; j < CK; ++j) mrow = fmaxf(mrow, sS[r * CK + j]);
            float m_old = sM[r];
            float m_new = fmaxf(m_old, mrow);
            float alpha = __expf(m_old - m_new);
            float lsum = 0.0f;
#pragma unroll
            for (int j = 0; j < CK; ++j) {
                float p = __expf(sS[r * CK + j] - m_new);
                sP[r * CK + j] = (_Float16)p;
                lsum += p;
            }
            sM[r]  = m_new;
            sL[r]  = sL[r] * alpha + lsum;
            sAl[r] = alpha;
        }
        __syncthreads();

        // ---- rescale O and accumulate P(64x32) @ V(32x128) ----
        {
#pragma unroll
            for (int e = 0; e < 8; ++e) {
                float al = sAl[wr * 16 + hi * 8 + e];
#pragma unroll
                for (int jt = 0; jt < 4; ++jt) oacc[jt][e] *= al;
            }
            v16h af;
            const _Float16* ap = &sP[(wr * 16 + row16) * CK + hi * 16];
#pragma unroll
            for (int j = 0; j < 8; ++j) af[j] = ap[j];
#pragma unroll
            for (int j = 0; j < 8; ++j) af[8 + j] = ap[8 + j];
#pragma unroll
            for (int jt = 0; jt < 4; ++jt) {
                v16h bf;   // B[kk][col] = V[key kk][hd col]
                const _Float16* bp = &sV[cur][(hi * 16) * HD_ + wc * 64 + jt * 16 + row16];
#pragma unroll
                for (int j = 0; j < 16; ++j) bf[j] = bp[j * HD_];
                oacc[jt] = __builtin_amdgcn_wmma_f32_16x16x32_f16(
                    false, af, false, bf, (short)0, oacc[jt], false, false);
            }
        }
        __syncthreads();   // protect sK/sV/sP before next chunk's TDM writes
    }

    // ---- normalize and write out [B, L, H*HD] f16 ----
#pragma unroll
    for (int e = 0; e < 8; ++e) {
        int rl = wr * 16 + hi * 8 + e;
        float inv = 1.0f / sL[rl];
#pragma unroll
        for (int jt = 0; jt < 4; ++jt) {
            int col = wc * 64 + jt * 16 + row16;
            out[((size_t)(b * L_ + q0 + rl)) * (size_t)D_ + h * HD_ + col] =
                (_Float16)(oacc[jt][e] * inv);
        }
    }
}

// ---------------- host-side launch ----------------
extern "C" void kernel_launch(void* const* d_in, const int* in_sizes, int n_in,
                              void* d_out, int out_size, void* d_ws, size_t ws_size,
                              hipStream_t stream) {
    (void)in_sizes; (void)n_in; (void)out_size; (void)ws_size;
    const float* x      = (const float*)d_in[0];
    const float* mask   = (const float*)d_in[1];
    const float* past_k = (const float*)d_in[2];
    const float* past_v = (const float*)d_in[3];
    const float* Wq     = (const float*)d_in[4];
    const float* Wk     = (const float*)d_in[5];
    const float* Wv     = (const float*)d_in[6];
    const float* Wo     = (const float*)d_in[7];
    float* outp = (float*)d_out;

    const size_t n_xD   = (size_t)B_ * L_ * D_;
    const size_t n_WqD  = (size_t)D_ * D_;
    const size_t n_Wk   = (size_t)D_ * KH_ * HD_;
    const size_t n_kv   = (size_t)B_ * KH_ * S_ * HD_;
    const size_t n_proj = (size_t)B_ * L_ * KH_ * HD_;
    const size_t n_past = (size_t)B_ * KH_ * P_ * HD_;

    char* ws = (char*)d_ws;
    auto take = [&](size_t halfs) { _Float16* p = (_Float16*)ws; ws += halfs * sizeof(_Float16); return p; };
    _Float16* x16    = take(n_xD);
    _Float16* Wq16   = take(n_WqD);
    _Float16* Wk16   = take(n_Wk);
    _Float16* Wv16   = take(n_Wk);
    _Float16* Wo16   = take(n_WqD);
    _Float16* q16    = take(n_xD);
    _Float16* kproj  = take(n_proj);
    _Float16* vproj  = take(n_proj);
    _Float16* k_all  = take(n_kv);
    _Float16* v_all  = take(n_kv);
    _Float16* attn16 = take(n_xD);

    auto cvt = [&](const float* s, _Float16* d, size_t n) {
        int n4 = (int)(n / 4);
        cvt_f32_to_f16<<<(n4 + 255) / 256, 256, 0, stream>>>(s, d, n4);
    };
    cvt(x,  x16,  n_xD);
    cvt(Wq, Wq16, n_WqD);
    cvt(Wk, Wk16, n_Wk);
    cvt(Wv, Wv16, n_Wk);
    cvt(Wo, Wo16, n_WqD);

    pack_past<<<((int)n_past + 255) / 256, 256, 0, stream>>>(past_k, k_all, (int)n_past);
    pack_past<<<((int)n_past + 255) / 256, 256, 0, stream>>>(past_v, v_all, (int)n_past);

    const int M = B_ * L_;   // 4096
    gemm_wmma_f16<_Float16><<<dim3(D_ / 128, M / 128), 256, 0, stream>>>(x16, Wq16, q16, M, D_, D_);
    gemm_wmma_f16<_Float16><<<dim3((KH_ * HD_) / 128, M / 128), 256, 0, stream>>>(x16, Wk16, kproj, M, KH_ * HD_, D_);
    gemm_wmma_f16<_Float16><<<dim3((KH_ * HD_) / 128, M / 128), 256, 0, stream>>>(x16, Wv16, vproj, M, KH_ * HD_, D_);

    pack_new<<<((int)n_proj + 255) / 256, 256, 0, stream>>>(kproj, k_all, (int)n_proj);
    pack_new<<<((int)n_proj + 255) / 256, 256, 0, stream>>>(vproj, v_all, (int)n_proj);

    attn_flash<<<dim3(B_ * H_, L_ / 64), 256, 0, stream>>>(q16, k_all, v_all, mask, attn16);

    gemm_wmma_f16<float><<<dim3(D_ / 128, M / 128), 256, 0, stream>>>(attn16, Wo16, outp, M, D_, D_);
}
